// DenseSegGNNBlock_82755429859956
// MI455X (gfx1250) — matrix operand order repaired and checked by reading, not compiled, not run on hardware
//
#include <hip/hip_runtime.h>
#include <hip/hip_bf16.h>

typedef __attribute__((ext_vector_type(16))) __bf16 v16bf;
typedef __attribute__((ext_vector_type(8)))  float  v8f;
typedef __attribute__((__vector_size__(8)))  int    vs2i;   // matches builtin param
typedef __attribute__((__vector_size__(16))) int    vs4i;   // matches builtin param

#define AS1 __attribute__((address_space(1)))
#define AS3 __attribute__((address_space(3)))

#if __has_builtin(__builtin_amdgcn_cvt_pk_bf16_f32)
#define HAVE_PK_BF16 1
#endif
#if __has_builtin(__builtin_amdgcn_global_load_async_to_lds_b64)
#define HAVE_ASYNC_B64 1
#endif
#if __has_builtin(__builtin_amdgcn_global_load_async_to_lds_b128)
#define HAVE_ASYNC_B128 1
#endif

union ABFrag {
    uint2 u2[4];
    v16bf v;
};

__device__ __forceinline__ unsigned pack_bf2(float lo, float hi) {
#ifdef HAVE_PK_BF16
    return __builtin_bit_cast(unsigned, __builtin_amdgcn_cvt_pk_bf16_f32(lo, hi));
#else
    union { float f; unsigned u; } a, b;
    a.f = lo; b.f = hi;
    unsigned ra = (a.u + 0x7FFFu + ((a.u >> 16) & 1u)) >> 16;
    unsigned rb = (b.u + 0x7FFFu + ((b.u >> 16) & 1u)) >> 16;
    return ra | (rb << 16);
#endif
}

__device__ __forceinline__ unsigned short f2bf(float f) {
#ifdef HAVE_PK_BF16
    return (unsigned short)(pack_bf2(f, f) & 0xffffu);
#else
    union { float f; unsigned u; } v; v.f = f;
    unsigned r = v.u + 0x7FFFu + ((v.u >> 16) & 1u);
    return (unsigned short)(r >> 16);
#endif
}

// stage 64 contiguous bytes (already-bf16 data) from global into LDS
__device__ __forceinline__ void stage_row_64B(const unsigned short* g, unsigned short* l) {
#ifdef HAVE_ASYNC_B64
    AS1 vs2i* gp = (AS1 vs2i*)const_cast<unsigned short*>(g);
    AS3 vs2i* lp = (AS3 vs2i*)l;
    __builtin_amdgcn_global_load_async_to_lds_b64(gp, lp, 0,  0);
    __builtin_amdgcn_global_load_async_to_lds_b64(gp, lp, 8,  0);
    __builtin_amdgcn_global_load_async_to_lds_b64(gp, lp, 16, 0);
    __builtin_amdgcn_global_load_async_to_lds_b64(gp, lp, 24, 0);
    __builtin_amdgcn_global_load_async_to_lds_b64(gp, lp, 32, 0);
    __builtin_amdgcn_global_load_async_to_lds_b64(gp, lp, 40, 0);
    __builtin_amdgcn_global_load_async_to_lds_b64(gp, lp, 48, 0);
    __builtin_amdgcn_global_load_async_to_lds_b64(gp, lp, 56, 0);
#else
    const uint4* sp = (const uint4*)g;
    uint4 q0 = sp[0], q1 = sp[1], q2 = sp[2], q3 = sp[3];
    uint2* bd = (uint2*)l;
    bd[0] = make_uint2(q0.x, q0.y); bd[1] = make_uint2(q0.z, q0.w);
    bd[2] = make_uint2(q1.x, q1.y); bd[3] = make_uint2(q1.z, q1.w);
    bd[4] = make_uint2(q2.x, q2.y); bd[5] = make_uint2(q2.z, q2.w);
    bd[6] = make_uint2(q3.x, q3.y); bd[7] = make_uint2(q3.z, q3.w);
#endif
}

// stage 128 contiguous bytes (16B aligned) from global into LDS
__device__ __forceinline__ void stage_row_128B(const unsigned short* g, unsigned short* l) {
#ifdef HAVE_ASYNC_B128
    AS1 vs4i* gp = (AS1 vs4i*)const_cast<unsigned short*>(g);
    AS3 vs4i* lp = (AS3 vs4i*)l;
    __builtin_amdgcn_global_load_async_to_lds_b128(gp, lp, 0,   0);
    __builtin_amdgcn_global_load_async_to_lds_b128(gp, lp, 16,  0);
    __builtin_amdgcn_global_load_async_to_lds_b128(gp, lp, 32,  0);
    __builtin_amdgcn_global_load_async_to_lds_b128(gp, lp, 48,  0);
    __builtin_amdgcn_global_load_async_to_lds_b128(gp, lp, 64,  0);
    __builtin_amdgcn_global_load_async_to_lds_b128(gp, lp, 80,  0);
    __builtin_amdgcn_global_load_async_to_lds_b128(gp, lp, 96,  0);
    __builtin_amdgcn_global_load_async_to_lds_b128(gp, lp, 112, 0);
#else
    const uint4* src = (const uint4*)g;
    uint4* dst = (uint4*)l;
    #pragma unroll
    for (int q = 0; q < 8; ++q) dst[q] = src[q];
#endif
}

__device__ __forceinline__ void wait_async0() {
#if defined(HAVE_ASYNC_B64) || defined(HAVE_ASYNC_B128)
#if __has_builtin(__builtin_amdgcn_s_wait_asynccnt)
    __builtin_amdgcn_s_wait_asynccnt(0);
#else
    asm volatile("s_wait_asynccnt 0" ::: "memory");
#endif
#endif
}

// ---- WMMA helpers (layouts per CDNA5 ISA 7.12.2, wave32) -------------------

// A fragment 16x32 bf16: lane m (0-15) holds row m, K = {kb..kb+7, kb+16..kb+23},
// kb = 0 for lanes 0-15, kb = 8 for lanes 16-31.
__device__ __forceinline__ v16bf load_frag_a(const unsigned short* base, int ld, int lane) {
    const int m  = lane & 15;
    const int kb = (lane >> 4) << 3;               // 0 or 8
    const unsigned short* p = base + m * ld + kb;
    ABFrag f;
    f.u2[0] = *(const uint2*)(p);
    f.u2[1] = *(const uint2*)(p + 4);
    f.u2[2] = *(const uint2*)(p + 16);
    f.u2[3] = *(const uint2*)(p + 20);
    return f.v;
}

// B fragment 32x16 bf16 (stored column-major in LDS: [col][k]):
// lane n (0-15) holds column n, K = {kb..kb+7, kb+16..kb+23}; kb = 0 / 8 per half-wave.
__device__ __forceinline__ v16bf load_frag_b(const unsigned short* base, int ld, int lane) {
    const int n  = lane & 15;
    const int kb = (lane >> 4) << 3;               // 0 or 8
    const unsigned short* p = base + n * ld + kb;
    ABFrag f;
    f.u2[0] = *(const uint2*)(p);
    f.u2[1] = *(const uint2*)(p + 4);
    f.u2[2] = *(const uint2*)(p + 16);
    f.u2[3] = *(const uint2*)(p + 20);
    return f.v;
}

__device__ __forceinline__ v8f wmma_bf16(v16bf a, v16bf b, v8f c) {
    // (neg_a, A, neg_b, B, c_mod, C, reuse_a, reuse_b)
    return __builtin_amdgcn_wmma_f32_16x16x32_bf16(false, a, false, b, (short)0, c, false, false);
}

// ---- Kernel 1: LayerNorm -> h_row (bf16, row-major) + hT (bf16, transposed)
__global__ __launch_bounds__(256) void gnn_ln_kernel(
    const float* __restrict__ x, const float* __restrict__ gamma,
    const float* __restrict__ beta,
    unsigned short* __restrict__ h_row, unsigned short* __restrict__ hT)
{
    const int lane = threadIdx.x & 31;
    const int wave = threadIdx.x >> 5;
    const int row  = blockIdx.x * 8 + wave;        // 0 .. 32767
    const float* xr = x + (size_t)row * 256 + lane * 8;
    float4 a = ((const float4*)xr)[0];
    float4 b = ((const float4*)xr)[1];

    float s = a.x + a.y + a.z + a.w + b.x + b.y + b.z + b.w;
    #pragma unroll
    for (int off = 16; off; off >>= 1) s += __shfl_xor(s, off, 32);
    const float mu = s * (1.0f / 256.0f);

    float d0 = a.x - mu, d1 = a.y - mu, d2 = a.z - mu, d3 = a.w - mu;
    float d4 = b.x - mu, d5 = b.y - mu, d6 = b.z - mu, d7 = b.w - mu;
    float vs = d0*d0 + d1*d1 + d2*d2 + d3*d3 + d4*d4 + d5*d5 + d6*d6 + d7*d7;
    #pragma unroll
    for (int off = 16; off; off >>= 1) vs += __shfl_xor(vs, off, 32);
    const float rstd = rsqrtf(vs * (1.0f / 256.0f) + 1e-5f);

    const float4 g0 = ((const float4*)(gamma + lane * 8))[0];
    const float4 g1 = ((const float4*)(gamma + lane * 8))[1];
    const float4 e0 = ((const float4*)(beta  + lane * 8))[0];
    const float4 e1 = ((const float4*)(beta  + lane * 8))[1];

    float y0 = d0 * rstd * g0.x + e0.x, y1 = d1 * rstd * g0.y + e0.y;
    float y2 = d2 * rstd * g0.z + e0.z, y3 = d3 * rstd * g0.w + e0.w;
    float y4 = d4 * rstd * g1.x + e1.x, y5 = d5 * rstd * g1.y + e1.y;
    float y6 = d6 * rstd * g1.z + e1.z, y7 = d7 * rstd * g1.w + e1.w;

    uint4 hv;
    hv.x = pack_bf2(y0, y1); hv.y = pack_bf2(y2, y3);
    hv.z = pack_bf2(y4, y5); hv.w = pack_bf2(y6, y7);
    *(uint4*)(h_row + (size_t)row * 256 + lane * 8) = hv;      // coalesced 16B

    const int bidx = row >> 10, k = row & 1023;
    unsigned short ys[8] = { f2bf(y0), f2bf(y1), f2bf(y2), f2bf(y3),
                             f2bf(y4), f2bf(y5), f2bf(y6), f2bf(y7) };
    unsigned short* ht = hT + ((size_t)bidx * 256 + lane * 8) * 1024 + k;
    #pragma unroll
    for (int e = 0; e < 8; ++e) ht[(size_t)e * 1024] = ys[e];  // L2 write-combines
}

// ---- Kernel 2: W (f32, HxH) -> WT (bf16, transposed: WT[o*H+h] = W[h*H+o])
__global__ __launch_bounds__(256) void gnn_wprep_kernel(
    const float* __restrict__ Wrel, const float* __restrict__ Wroot,
    unsigned short* __restrict__ WTrel, unsigned short* __restrict__ WTroot)
{
    const int i = blockIdx.x * 256 + threadIdx.x;              // 0 .. 131071
    const float* W        = (i < 65536) ? Wrel  : Wroot;
    unsigned short* WT    = (i < 65536) ? WTrel : WTroot;
    const int j = i & 65535;
    const int o = j >> 8, h = j & 255;
    WT[j] = f2bf(W[h * 256 + o]);
}

// ---- Kernel 3: fused  agg = adj@h ; conv = agg@W_rel + h@W_root + b ; out = x + relu(conv)
#define BM  64
#define BK  32
#define LDA 36     // padded bf16 stride (72B rows -> 8B aligned, conflict-friendly)
#define LDB 36
#define LDC 264    // 528B rows -> 16B aligned

__global__ __launch_bounds__(256) void gnn_gemm_kernel(
    const float* __restrict__ x, const float* __restrict__ adj,
    const float* __restrict__ b_rel,
    const unsigned short* __restrict__ hT,
    const unsigned short* __restrict__ h_row,
    const unsigned short* __restrict__ WTrel,
    const unsigned short* __restrict__ WTroot,
    float* __restrict__ out)
{
    __shared__ unsigned short As[BM * LDA];        //  4608 B
    __shared__ unsigned short Bs[256 * LDB];       // 18432 B
    __shared__ unsigned short Cs[BM * LDC];        // 33792 B  (agg tile, then h tile)

    const int tid  = threadIdx.x;
    const int lane = tid & 31;
    const int wave = tid >> 5;
    const int wm   = wave >> 2;                    // 0..1  -> 32-row strip
    const int wn   = wave & 3;                     // 0..3  -> 64-col strip
    const int bb   = blockIdx.y;                   // batch
    const int m0   = blockIdx.x * BM;              // row tile

    const v8f vzero = {0.f,0.f,0.f,0.f,0.f,0.f,0.f,0.f};
    v8f c1[2][4];
    #pragma unroll
    for (int i = 0; i < 2; ++i)
        #pragma unroll
        for (int j = 0; j < 4; ++j) c1[i][j] = vzero;

    // ---------------- Phase 1: agg tile = adj[m0:m0+64, :] @ h  -------------
    const int ar = tid >> 2;                       // 0..63
    const int ac = (tid & 3) * 8;                  // 0,8,16,24
    const size_t adj_base = ((size_t)bb * 1024 + m0 + ar) * 1024 + ac;
    const unsigned short* hT_b = hT + ((size_t)bb * 256 + tid) * 1024;

    for (int kk = 0; kk < 1024; kk += BK) {
        // stage B: hT (already transposed, bf16) -> Bs[col][k]  (async if avail)
        stage_row_64B(hT_b + kk, &Bs[tid * LDB]);

        // stage A: adj f32 -> bf16 (each thread: 8 contiguous elems of one row)
        const float* ap = adj + adj_base + kk;
        __builtin_prefetch(ap + 64, 0, 1);         // speculative global_prefetch_b8
        float4 f0 = ((const float4*)ap)[0];
        float4 f1 = ((const float4*)ap)[1];
        uint2 s0, s1;
        s0.x = pack_bf2(f0.x, f0.y); s0.y = pack_bf2(f0.z, f0.w);
        s1.x = pack_bf2(f1.x, f1.y); s1.y = pack_bf2(f1.z, f1.w);
        uint2* adst = (uint2*)&As[ar * LDA + ac];
        adst[0] = s0; adst[1] = s1;

        wait_async0();
        __syncthreads();

        ABFrag afr[2], bfr[4];
        #pragma unroll
        for (int i = 0; i < 2; ++i)
            afr[i].v = load_frag_a(&As[(wm * 32 + i * 16) * LDA], LDA, lane);
        #pragma unroll
        for (int j = 0; j < 4; ++j)
            bfr[j].v = load_frag_b(&Bs[(wn * 64 + j * 16) * LDB], LDB, lane);
        #pragma unroll
        for (int i = 0; i < 2; ++i)
            #pragma unroll
            for (int j = 0; j < 4; ++j)
                c1[i][j] = wmma_bf16(afr[i].v, bfr[j].v, c1[i][j]);
        __syncthreads();
    }

    // spill agg tile (bf16) into Cs  (C layout: VGPR g -> M = g (+8 upper half))
    {
        const int n0    = wn * 64 + (lane & 15);
        const int rbase = wm * 32 + ((lane >> 4) << 3);
        #pragma unroll
        for (int i = 0; i < 2; ++i)
            #pragma unroll
            for (int j = 0; j < 4; ++j)
                #pragma unroll
                for (int g = 0; g < 8; ++g)
                    Cs[(rbase + i * 16 + g) * LDC + n0 + j * 16] = f2bf(c1[i][j][g]);
    }
    __syncthreads();

    // ---------------- Phase 2: conv = agg@W_rel + h@W_root ------------------
    v8f c2[2][4];
    #pragma unroll
    for (int i = 0; i < 2; ++i)
        #pragma unroll
        for (int j = 0; j < 4; ++j) c2[i][j] = vzero;

    for (int p = 0; p < 2; ++p) {
        const unsigned short* WT = p ? WTroot : WTrel;
        if (p == 1) {
            // overwrite Cs with h rows (L2-resident) for the W_root pass
            const int r  = tid >> 2;
            const int c0 = (tid & 3) * 64;
            stage_row_128B(h_row + ((size_t)bb * 1024 + m0 + r) * 256 + c0,
                           &Cs[r * LDC + c0]);
            wait_async0();
            __syncthreads();
        }
        for (int k2 = 0; k2 < 256; k2 += BK) {
            // stage Bs from transposed weights (bf16, async if avail)
            stage_row_64B(WT + (size_t)tid * 256 + k2, &Bs[tid * LDB]);
            wait_async0();
            __syncthreads();

            ABFrag afr[2], bfr[4];
            #pragma unroll
            for (int i = 0; i < 2; ++i)
                afr[i].v = load_frag_a(&Cs[(wm * 32 + i * 16) * LDC + k2], LDC, lane);
            #pragma unroll
            for (int j = 0; j < 4; ++j)
                bfr[j].v = load_frag_b(&Bs[(wn * 64 + j * 16) * LDB], LDB, lane);
            #pragma unroll
            for (int i = 0; i < 2; ++i)
                #pragma unroll
                for (int j = 0; j < 4; ++j)
                    c2[i][j] = wmma_bf16(afr[i].v, bfr[j].v, c2[i][j]);
            __syncthreads();
        }
    }

    // ---------------- Phase 3: out = x + relu(conv + b_rel) -----------------
    {
        const int n0 = wn * 64 + (lane & 15);
        const int rb = wm * 32 + ((lane >> 4) << 3);
        #pragma unroll
        for (int j = 0; j < 4; ++j) {
            const int col = n0 + j * 16;
            const float br = b_rel[col];
            #pragma unroll
            for (int i = 0; i < 2; ++i) {
                #pragma unroll
                for (int g = 0; g < 8; ++g) {
                    const size_t idx =
                        ((size_t)bb * 1024 + m0 + rb + i * 16 + g) * 256 + col;
                    float v = c2[i][j][g] + br;
                    v = v > 0.f ? v : 0.f;
                    out[idx] = x[idx] + v;
                }
            }
        }
    }
}

extern "C" void kernel_launch(void* const* d_in, const int* in_sizes, int n_in,
                              void* d_out, int out_size, void* d_ws, size_t ws_size,
                              hipStream_t stream) {
    const float* x      = (const float*)d_in[0];
    const float* adj    = (const float*)d_in[1];
    const float* W_rel  = (const float*)d_in[2];
    const float* b_rel  = (const float*)d_in[3];
    const float* W_root = (const float*)d_in[4];
    const float* gamma  = (const float*)d_in[5];
    const float* beta   = (const float*)d_in[6];
    float* out = (float*)d_out;

    // workspace layout (bf16): hT (16MB) | h_row (16MB) | WT_rel | WT_root
    unsigned short* hT     = (unsigned short*)d_ws;
    unsigned short* h_row  = hT    + (size_t)32 * 256 * 1024;
    unsigned short* WTrel  = h_row + (size_t)32 * 1024 * 256;
    unsigned short* WTroot = WTrel + 256 * 256;

    gnn_ln_kernel<<<4096, 256, 0, stream>>>(x, gamma, beta, h_row, hT);
    gnn_wprep_kernel<<<512, 256, 0, stream>>>(W_rel, W_root, WTrel, WTroot);
    gnn_gemm_kernel<<<dim3(16, 32), 256, 0, stream>>>(
        x, adj, b_rel, hT, h_row, WTrel, WTroot, out);
}
